// Refine_35845797053278
// MI455X (gfx1250) — compile-verified, matrix operand order checked by reading, not run on hardware
//
#include <hip/hip_runtime.h>
#include <hip/hip_bf16.h>

typedef __attribute__((ext_vector_type(2)))  float  v2f;
typedef __attribute__((ext_vector_type(8)))  float  v8f;
typedef __attribute__((ext_vector_type(16))) __bf16 v16bf;

#define C_DIM 256
#define HW    16384          // 128*128
#define NCENT 80
#define NB    8
#define PIX_TOTAL (NB * HW)  // 131072

// ---- workspace layout (byte offsets) ----
#define WS_INV    0                        // 80 f32 centroid inverse norms
#define WS_CFRAG  1024                     // f32 WMMA B-frags of norm. centroids: 64 kt * 5 nt * 32 lanes * 2 f32 = 81920 B
#define WS_WFRAG  (WS_CFRAG + 81920)       // bf16 WMMA B-frags of fc_w: [ot=16][kt=8][lane=32] * 32 B = 131072 B
#define WS_ASSIGN (WS_WFRAG + 131072)      // 131072 int32
#define WS_COUNT  (WS_ASSIGN + 524288)     // 640 f32 (padded to 4096 B)
#define WS_SUM    (WS_COUNT + 4096)        // 8*80*256 f32 = 655360 B (becomes local_cent in place)

__device__ __forceinline__ __bf16 f2bf(float x) { return (__bf16)x; }

// ---------------- Kernel 1a: centroid inverse L2 norms ----------------
__global__ void k_cnorm(const float* __restrict__ cent, float* __restrict__ inv) {
    __shared__ float red[256];
    int t = threadIdx.x;
    float v = cent[blockIdx.x * C_DIM + t];
    red[t] = v * v;
    __syncthreads();
    for (int s = 128; s > 0; s >>= 1) {
        if (t < s) red[t] += red[t + s];
        __syncthreads();
    }
    if (t == 0) inv[blockIdx.x] = 1.0f / fmaxf(sqrtf(red[0]), 1e-12f);
}

// ---------------- Kernel 1b: normalized-centroid f32 B-fragments ----------------
// B frag for V_WMMA_F32_16X16X4_F32: lane l holds B[k][n], n = l%16, k = kb + 2*(l/16) + j  (j=0,1)
__global__ void k_cfrag(const float* __restrict__ cent, const float* __restrict__ inv,
                        float* __restrict__ cf) {
    int tid = blockIdx.x * blockDim.x + threadIdx.x;   // 10240 = 64*5*32
    int kt   = tid / 160;
    int rem  = tid % 160;
    int t    = rem / 32;
    int lane = rem % 32;
    int n = t * 16 + (lane & 15);
    int k = kt * 4 + 2 * (lane >> 4);
    float s = inv[n];
    cf[tid * 2 + 0] = cent[n * C_DIM + k + 0] * s;
    cf[tid * 2 + 1] = cent[n * C_DIM + k + 1] * s;
}

// ---------------- Kernel 1c: fc_w bf16 B-fragments, [ot][kt][lane] layout ----------------
// B frag for V_WMMA_F32_16X16X32_BF16: lane l, n=l%16, h=l/16; dword d: K0 = kt*32 + (d&4?16:0) + 8h + (d&3)*2
__global__ void k_wfrag(const float* __restrict__ fcw, unsigned int* __restrict__ wf) {
    int tid = blockIdx.x * blockDim.x + threadIdx.x;   // 4096 = 16*8*32, ot-major
    int ot   = tid >> 8;
    int kt   = (tid >> 5) & 7;
    int lane = tid & 31;
    int o = ot * 16 + (lane & 15);
    int h = lane >> 4;
#pragma unroll
    for (int d = 0; d < 8; ++d) {
        int K0 = kt * 32 + ((d & 4) ? 16 : 0) + 8 * h + (d & 3) * 2;
        unsigned int lo = (unsigned int)__builtin_bit_cast(unsigned short, f2bf(fcw[o * C_DIM + K0]));
        unsigned int hi = (unsigned int)__builtin_bit_cast(unsigned short, f2bf(fcw[o * C_DIM + K0 + 1]));
        wf[tid * 8 + d] = lo | (hi << 16);
    }
}

// ---------------- Kernel 2: cosine-sim GEMM (f32 WMMA) + argmax -> assign ----------------
__global__ void k_assign(const float* __restrict__ feat, const float* __restrict__ cf,
                         int* __restrict__ assign) {
    __shared__ float sims[8][16][81];                  // 81 stride kills bank conflicts
    int wave = threadIdx.x >> 5;
    int lane = threadIdx.x & 31;
    int m = lane & 15;
    int h = lane >> 4;
    int gp0 = blockIdx.x * 128 + wave * 16;            // global pixel base for this wave
    int b = gp0 >> 14;
    int p0 = gp0 & (HW - 1);
    const float* fbase = feat + (size_t)b * C_DIM * HW + p0 + m;

    v8f acc[5];
#pragma unroll
    for (int t = 0; t < 5; ++t)
#pragma unroll
        for (int r = 0; r < 8; ++r) acc[t][r] = 0.0f;

    const v2f* cf2 = (const v2f*)cf;
    for (int kt = 0; kt < 64; ++kt) {
        int kb = kt * 4 + 2 * h;
        v2f av;
        av.x = fbase[(size_t)(kb + 0) * HW];
        av.y = fbase[(size_t)(kb + 1) * HW];
#pragma unroll
        for (int t = 0; t < 5; ++t) {
            v2f bv = cf2[(kt * 5 + t) * 32 + lane];
            acc[t] = __builtin_amdgcn_wmma_f32_16x16x4_f32(
                false, av, false, bv, (short)0, acc[t], false, false);
        }
    }
    // spill sims to LDS: lane holds (M = r + 8h, N = t*16 + lane%16)
#pragma unroll
    for (int t = 0; t < 5; ++t)
#pragma unroll
        for (int r = 0; r < 8; ++r)
            sims[wave][r + 8 * h][t * 16 + m] = acc[t][r];
    __syncthreads();
    if (lane < 16) {
        float best = -3.4e38f;
        int bi = 0;
        for (int n = 0; n < NCENT; ++n) {
            float v = sims[wave][lane][n];
            if (v > best) { best = v; bi = n; }       // strict > == first-max (jnp.argmax)
        }
        assign[gp0 + lane] = bi;
    }
}

// ---------------- Kernel 3a: zero count+sum ----------------
__global__ void k_zero(float* __restrict__ p, int n) {
    int i = blockIdx.x * blockDim.x + threadIdx.x;
    if (i < n) p[i] = 0.0f;
}

// ---------------- Kernel 3b: scatter via LDS accumulation (ds_add_f32), then flush ----------------
// grid: 128 blocks = [b:8][half:2][pchunk:8]; block = 256 threads covers 2048 pixels x 128 channels
__global__ void k_scatter(const float* __restrict__ feat, const int* __restrict__ assign,
                          float* __restrict__ count, float* __restrict__ sum) {
    __shared__ float accs[NCENT * 129];                // 129 pitch: lanes w/ different centroid -> different banks
    __shared__ float cnt[NCENT];
    int t = threadIdx.x;
    int blk = blockIdx.x;
    int b    = blk >> 4;
    int half = (blk >> 3) & 1;
    int pc   = blk & 7;
    int pbase = pc * 2048;

    for (int i = t; i < NCENT * 129; i += 256) accs[i] = 0.0f;
    if (t < NCENT) cnt[t] = 0.0f;
    __syncthreads();

    int ap[8];
#pragma unroll
    for (int i = 0; i < 8; ++i) ap[i] = assign[b * HW + pbase + t + 256 * i];
    if (half == 0) {
#pragma unroll
        for (int i = 0; i < 8; ++i) atomicAdd(&cnt[ap[i]], 1.0f);
    }
    const float* fb = feat + (size_t)b * C_DIM * HW + pbase + t;
    for (int c = 0; c < 128; ++c) {
        const float* row = fb + (size_t)(half * 128 + c) * HW;
#pragma unroll
        for (int i = 0; i < 8; ++i)
            atomicAdd(&accs[ap[i] * 129 + c], row[256 * i]);   // LDS atomic
    }
    __syncthreads();
    // flush partial sums to global (one atomic per (n,c) per block)
    for (int i = t; i < NCENT * 128; i += 256) {
        int n = i >> 7;
        int c = i & 127;
        atomicAdd(&sum[((size_t)(b * NCENT + n)) * C_DIM + half * 128 + c], accs[n * 129 + c]);
    }
    if (half == 0 && t < NCENT) atomicAdd(&count[b * NCENT + t], cnt[t]);
}

// ---------------- Kernel 4: local_cent = sum / max(count,1) (in place) ----------------
__global__ void k_div(float* __restrict__ sum, const float* __restrict__ count) {
    int i = blockIdx.x * blockDim.x + threadIdx.x;     // 163840
    float cnt = count[i >> 8];
    sum[i] = sum[i] / fmaxf(cnt, 1.0f);
}

// ---------------- Kernel 5: fused delta/alpha/x + 1x1 conv (bf16 WMMA) + bias + ReLU ----------------
__global__ void k_conv(const float* __restrict__ feat, const int* __restrict__ assign,
                       const float* __restrict__ lcent, const unsigned int* __restrict__ wfraw,
                       const float* __restrict__ fcb, float* __restrict__ out) {
    __shared__ __align__(16) unsigned int wlds[2048];  // 8 KB: one ot-chunk of B-fragments
    __shared__ float tr[4][16][17];                    // 17 stride: conflict-free transpose
    int tid  = threadIdx.x;
    int wave = tid >> 5;
    int lane = tid & 31;
    int m = lane & 15;
    int h = lane >> 4;
    int gp0 = blockIdx.x * 64 + wave * 16;
    int b = gp0 >> 14;
    int p0 = gp0 & (HW - 1);
    const float* fpix = feat + (size_t)b * C_DIM * HW + p0 + m;
    int a = assign[gp0 + m];
    const float* cent = lcent + ((size_t)(b * NCENT + a)) * C_DIM;

    // phase 1: alpha = exp(-mean_c(delta^2)); each lane sums 128 channels of its pixel
    float ss = 0.0f;
    int c0 = h * 128;
    for (int c = c0; c < c0 + 128; ++c) {
        float f = fpix[(size_t)c * HW];
        float d = cent[c] - f;
        ss += d * d;
    }
    float tot = ss + __shfl_xor(ss, 16, 32);           // wave32 cross-half reduce
    float alpha = __expf(-tot * (1.0f / 256.0f));

    // phase 2: build bf16 A-fragments of x = f + alpha*(cent-f), ISA layout
    v16bf af[8];
#pragma unroll
    for (int kt = 0; kt < 8; ++kt) {
#pragma unroll
        for (int d = 0; d < 8; ++d) {
            int K0 = kt * 32 + ((d & 4) ? 16 : 0) + 8 * h + (d & 3) * 2;
            float f0 = fpix[(size_t)K0 * HW];
            float f1 = fpix[(size_t)(K0 + 1) * HW];
            float x0 = f0 + alpha * (cent[K0] - f0);
            float x1 = f1 + alpha * (cent[K0 + 1] - f1);
            af[kt][2 * d + 0] = f2bf(x0);
            af[kt][2 * d + 1] = f2bf(x1);
        }
    }

    unsigned int lds_base = (unsigned int)(size_t)(&wlds[0]) + (unsigned int)tid * 64u;
    for (int ot = 0; ot < 16; ++ot) {
        // async-stage this ot's 8 KB B-fragment chunk into LDS (each of 128 threads moves 64 B)
        const char* gsrc = (const char*)(wfraw + (size_t)ot * 2048) + (size_t)tid * 64;
        asm volatile("global_load_async_to_lds_b128 %0, %1, off offset:0"  :: "v"(lds_base), "v"(gsrc) : "memory");
        asm volatile("global_load_async_to_lds_b128 %0, %1, off offset:16" :: "v"(lds_base), "v"(gsrc) : "memory");
        asm volatile("global_load_async_to_lds_b128 %0, %1, off offset:32" :: "v"(lds_base), "v"(gsrc) : "memory");
        asm volatile("global_load_async_to_lds_b128 %0, %1, off offset:48" :: "v"(lds_base), "v"(gsrc) : "memory");
        if (ot < 15) __builtin_prefetch((const char*)(wfraw + (size_t)(ot + 1) * 2048) + (size_t)tid * 64, 0, 0);
        asm volatile("s_wait_asynccnt 0" ::: "memory");
        __syncthreads();

        float bias = fcb[ot * 16 + m];
        v8f acc;
#pragma unroll
        for (int r = 0; r < 8; ++r) acc[r] = bias;
#pragma unroll
        for (int kt = 0; kt < 8; ++kt) {
            v16bf bw = *(const v16bf*)&wlds[(kt * 32 + lane) * 8];
            acc = __builtin_amdgcn_wmma_f32_16x16x32_bf16(
                false, af[kt], false, bw, (short)0, acc, false, false);
        }
        // ReLU + transpose through LDS so stores are pixel-contiguous
#pragma unroll
        for (int r = 0; r < 8; ++r)
            tr[wave][m][r + 8 * h] = fmaxf(acc[r], 0.0f);  // tr[N][M]
        __syncthreads();
        float* obase = out + ((size_t)b * C_DIM + ot * 16) * HW + p0;
#pragma unroll
        for (int i = 0; i < 8; ++i) {
            int ol = h + 2 * i;
            obase[(size_t)ol * HW + m] = tr[wave][ol][m];
        }
        __syncthreads();                                 // protect tr and wlds before next ot
    }
}

extern "C" void kernel_launch(void* const* d_in, const int* in_sizes, int n_in,
                              void* d_out, int out_size, void* d_ws, size_t ws_size,
                              hipStream_t stream) {
    const float* feat = (const float*)d_in[0];
    const float* cent = (const float*)d_in[1];
    const float* fcw  = (const float*)d_in[2];
    const float* fcb  = (const float*)d_in[3];
    float* outp = (float*)d_out;

    char* ws = (char*)d_ws;
    float*        inv    = (float*)(ws + WS_INV);
    float*        cfrag  = (float*)(ws + WS_CFRAG);
    unsigned int* wfrag  = (unsigned int*)(ws + WS_WFRAG);
    int*          assign = (int*)(ws + WS_ASSIGN);
    float*        count  = (float*)(ws + WS_COUNT);
    float*        sum    = (float*)(ws + WS_SUM);

    k_cnorm<<<NCENT, 256, 0, stream>>>(cent, inv);
    k_cfrag<<<40, 256, 0, stream>>>(cent, inv, cfrag);
    k_wfrag<<<16, 256, 0, stream>>>(fcw, wfrag);
    k_assign<<<PIX_TOTAL / 128, 256, 0, stream>>>(feat, cfrag, assign);
    int nz = (4096 + NB * NCENT * C_DIM * 4) / 4;      // count pad + sum, in floats
    k_zero<<<(nz + 255) / 256, 256, 0, stream>>>(count, nz);
    k_scatter<<<128, 256, 0, stream>>>(feat, assign, count, sum);
    k_div<<<(NB * NCENT * C_DIM) / 256, 256, 0, stream>>>(sum, count);
    k_conv<<<PIX_TOTAL / 64, 128, 0, stream>>>(feat, assign, sum, wfrag, fcb, outp);
}
